// DegradationMemory_25512105738422
// MI455X (gfx1250) — matrix operand order repaired and checked by reading, not compiled, not run on hardware
//
#include <hip/hip_runtime.h>
#include <hip/hip_bf16.h>

typedef __attribute__((ext_vector_type(16))) __bf16       v16bf;
typedef __attribute__((ext_vector_type(8)))  __bf16       v8bf;
typedef __attribute__((ext_vector_type(8)))  float        v8f;

#define TOUT 27
#define HH   64
#define WW   64
#define HWSZ 4096
#define NTOT (TOUT * HWSZ)   // 110592 output spatial positions

// ---------------------------------------------------------------------------
// prep: BN(eval)+bias folded into per-channel scale/shift; pw weights -> bf16
// ---------------------------------------------------------------------------
__global__ void prep_kernel(const float* __restrict__ pw_w, const float* __restrict__ pw_b,
                            const float* __restrict__ g,    const float* __restrict__ beta,
                            const float* __restrict__ mean, const float* __restrict__ var,
                            __bf16* __restrict__ wt, float* __restrict__ scale,
                            float* __restrict__ shift, int C) {
  int tid    = blockIdx.x * blockDim.x + threadIdx.x;
  int stride = gridDim.x * blockDim.x;
  if (tid < C) {
    float s    = g[tid] * rsqrtf(var[tid] + 1e-5f);
    scale[tid] = s;
    shift[tid] = (pw_b[tid] - mean[tid]) * s + beta[tid];
  }
  for (int i = tid; i < C * C; i += stride) wt[i] = (__bf16)pw_w[i];
}

// ---------------------------------------------------------------------------
// Depthwise 6x3x3 conv (valid over T, pad 1 over H/W) with perm gather on T.
// Output layout: [n = t*4096+h*64+w][C] bf16 (N x K row-major for GEMM B side)
// Block (16,16): x = w (coalesced fp32 reads), y = c; 16x17 LDS transpose so
// bf16 stores are contiguous over channels.
// ---------------------------------------------------------------------------
template <int C>
__global__ void dw_kernel(const float* __restrict__ x, const int* __restrict__ perm,
                          const float* __restrict__ dww, const float* __restrict__ dwb,
                          __bf16* __restrict__ out) {
  __shared__ float tile[16][17];
  const int w0 = blockIdx.x * 16;
  const int h  = blockIdx.y % HH;
  const int t  = blockIdx.y / HH;       // 0..26
  const int c0 = blockIdx.z * 16;
  const int wi = threadIdx.x;
  const int ci = threadIdx.y;
  const int c  = c0 + ci;
  const int w  = w0 + wi;

  float wreg[54];
#pragma unroll
  for (int i = 0; i < 54; ++i) wreg[i] = dww[c * 54 + i];

  float acc = dwb[c];
#pragma unroll
  for (int kt = 0; kt < 6; ++kt) {
    const int tsrc = perm[t + kt];
    const float* base = x + ((size_t)tsrc * C + c) * (size_t)HWSZ;
#pragma unroll
    for (int dh = 0; dh < 3; ++dh) {
      const int hi = h + dh - 1;
      if (hi < 0 || hi >= HH) continue;
#pragma unroll
      for (int dw = 0; dw < 3; ++dw) {
        const int wj = w + dw - 1;
        if (wj < 0 || wj >= WW) continue;
        acc += base[hi * WW + wj] * wreg[kt * 9 + dh * 3 + dw];
      }
    }
  }

  tile[ci][wi] = acc;                    // [channel][w]
  __syncthreads();
  const float v = tile[threadIdx.x][threadIdx.y];   // roles swapped
  const int n  = t * HWSZ + h * WW + (w0 + threadIdx.y);
  out[(size_t)n * C + c0 + threadIdx.x] = (__bf16)v;
}

// ---------------------------------------------------------------------------
// Pointwise GEMM + BN/ReLU via V_WMMA_F32_16X16X32_BF16, with double-buffered
// async global->LDS staging of the B (activation) K-tiles.
//   D[m, n] = sum_k Wt[m, k] * Bin[n, k]   (Bin stored N x K row-major)
// Block tile: MBLK(M) x NBLK(N); NWAVES = MBLK/16 waves, each owning 16 M rows
// and iterating NBLK/16 WMMA column tiles. K step 32.
// LDS rows padded to 48 bf16 (96B stride) -> 32B-aligned fragments, reduced
// bank conflicts. Generic->LDS address truncation per aperture rule.
// ---------------------------------------------------------------------------
template <int C, int MBLK, int NBLK>
__global__ void pw_gemm_kernel(const __bf16* __restrict__ Bin,  // [NTOT][C]
                               const __bf16* __restrict__ Wt,   // [C][C]
                               const float* __restrict__ scale,
                               const float* __restrict__ shift,
                               float* __restrict__ out) {
  constexpr int NWAVES   = MBLK / 16;
  constexpr int NTHREADS = NWAVES * 32;
  constexpr int NTILES   = NBLK / 16;
  constexpr int NCHUNKS  = NBLK * 4;          // 16B chunks per K-tile
  __shared__ __align__(64) __bf16 sB[2][NBLK][48];

  const int lane = threadIdx.x & 31;
  const int wave = threadIdx.x >> 5;
  const int l15  = lane & 15;
  const int lh   = lane >> 4;
  const int n_base = blockIdx.x * NBLK;
  const int m_base = blockIdx.y * MBLK + wave * 16;

  // async-copy one K-tile (NBLK rows x 32 bf16 = 64B/row) into sB[buf]
  auto stage_async = [&](int buf, int k) {
#pragma unroll
    for (int idx = threadIdx.x; idx < NCHUNKS; idx += NTHREADS) {
      const int row = idx >> 2, ch = idx & 3;
      const unsigned lds_off =
          (unsigned)(unsigned long long)&sB[buf][row][ch * 8];
      const unsigned long long gaddr =
          (unsigned long long)(Bin + (size_t)(n_base + row) * C + k + ch * 8);
      asm volatile("global_load_async_to_lds_b128 %0, %1, off"
                   :: "v"(lds_off), "v"(gaddr)
                   : "memory");
    }
  };

  stage_async(0, 0);

  v8f acc[NTILES] = {};
  int buf = 0;
  for (int k = 0; k < C; k += 32) {
    // wait for the tile we are about to consume (only outstanding asyncs),
    // then block-barrier so every wave sees the full LDS tile.
    asm volatile("s_wait_asynccnt 0x0" ::: "memory");
    __syncthreads();
    if (k + 32 < C) stage_async(buf ^ 1, k + 32);   // overlap with WMMAs

    // A fragment: lane<16 -> row m, K {0..15}; lane>=16 -> K {16..31}
    const __bf16* ap = Wt + (size_t)(m_base + l15) * C + k + 16 * lh;
    const v8bf alo = *(const v8bf*)ap;
    const v8bf ahi = *(const v8bf*)(ap + 8);
    v16bf a;
#pragma unroll
    for (int i = 0; i < 8; ++i) { a[i] = alo[i]; a[i + 8] = ahi[i]; }

#pragma unroll
    for (int j = 0; j < NTILES; ++j) {
      // B fragment: lane holds column n = n_base+j*16+l15, 16 contiguous K
      const v16bf b = *(const v16bf*)&sB[buf][j * 16 + l15][lh * 16];
      acc[j] = __builtin_amdgcn_wmma_f32_16x16x32_bf16(
          false, a, false, b, (short)0, acc[j], false, false);
    }
    buf ^= 1;
  }

  // epilogue: D lane layout -> lane l, vgpr r holds (m = 8*(l>>4)+r, n = l&15)
  float sc[8], sh[8];
#pragma unroll
  for (int r = 0; r < 8; ++r) {
    const int m = m_base + lh * 8 + r;
    sc[r] = scale[m];
    sh[r] = shift[m];
  }
#pragma unroll
  for (int j = 0; j < NTILES; ++j) {
    const int n  = n_base + j * 16 + l15;
    const int t  = n >> 12;        // / 4096
    const int hw = n & 4095;
    float* op = out + (size_t)t * C * HWSZ + hw;
#pragma unroll
    for (int r = 0; r < 8; ++r) {
      const int m = m_base + lh * 8 + r;
      op[(size_t)m * HWSZ] = fmaxf(acc[j][r] * sc[r] + sh[r], 0.0f);
    }
  }
}

// ---------------------------------------------------------------------------
extern "C" void kernel_launch(void* const* d_in, const int* in_sizes, int n_in,
                              void* d_out, int out_size, void* d_ws, size_t ws_size,
                              hipStream_t stream) {
  (void)in_sizes; (void)n_in; (void)out_size; (void)ws_size;

  const float* bank_key   = (const float*)d_in[0];
  const float* bank_value = (const float*)d_in[1];
  const int*   perm       = (const int*)d_in[2];
  const float* dw_k_w = (const float*)d_in[3];
  const float* dw_k_b = (const float*)d_in[4];
  const float* pw_k_w = (const float*)d_in[5];
  const float* pw_k_b = (const float*)d_in[6];
  const float* bn_k_g = (const float*)d_in[7];
  const float* bn_k_b = (const float*)d_in[8];
  const float* bn_k_m = (const float*)d_in[9];
  const float* bn_k_v = (const float*)d_in[10];
  const float* dw_v_w = (const float*)d_in[11];
  const float* dw_v_b = (const float*)d_in[12];
  const float* pw_v_w = (const float*)d_in[13];
  const float* pw_v_b = (const float*)d_in[14];
  const float* bn_v_g = (const float*)d_in[15];
  const float* bn_v_b = (const float*)d_in[16];
  const float* bn_v_m = (const float*)d_in[17];
  const float* bn_v_v = (const float*)d_in[18];

  float* out_key = (float*)d_out;                              // [27,64,64,64]
  float* out_val = out_key + (size_t)TOUT * 64 * HWSZ;         // [27,512,64,64]

  char* ws = (char*)d_ws;
  auto alloc = [&](size_t bytes) {
    char* p = ws;
    ws += (bytes + 255) & ~(size_t)255;
    return p;
  };
  __bf16* wt_k  = (__bf16*)alloc((size_t)64 * 64 * 2);
  __bf16* wt_v  = (__bf16*)alloc((size_t)512 * 512 * 2);
  float*  sc_k  = (float*)alloc(64 * 4);
  float*  sh_k  = (float*)alloc(64 * 4);
  float*  sc_v  = (float*)alloc(512 * 4);
  float*  sh_v  = (float*)alloc(512 * 4);
  __bf16* dwo_k = (__bf16*)alloc((size_t)NTOT * 64 * 2);       // ~14 MB
  __bf16* dwo_v = (__bf16*)alloc((size_t)NTOT * 512 * 2);      // ~113 MB

  prep_kernel<<<64, 256, 0, stream>>>(pw_k_w, pw_k_b, bn_k_g, bn_k_b, bn_k_m,
                                      bn_k_v, wt_k, sc_k, sh_k, 64);
  prep_kernel<<<256, 256, 0, stream>>>(pw_v_w, pw_v_b, bn_v_g, bn_v_b, bn_v_m,
                                       bn_v_v, wt_v, sc_v, sh_v, 512);

  dim3 dwb(16, 16);
  dw_kernel<64><<<dim3(4, HH * TOUT, 4), dwb, 0, stream>>>(bank_key, perm,
                                                           dw_k_w, dw_k_b, dwo_k);
  dw_kernel<512><<<dim3(4, HH * TOUT, 32), dwb, 0, stream>>>(bank_value, perm,
                                                             dw_v_w, dw_v_b, dwo_v);

  // key:   M=64,  K=64,  N=110592 -> 64M x 64N block tile,  4 waves
  // value: M=512, K=512, N=110592 -> 128M x 128N block tile, 8 waves
  pw_gemm_kernel<64, 64, 64><<<dim3(NTOT / 64, 1), 128, 0, stream>>>(
      dwo_k, wt_k, sc_k, sh_k, out_key);
  pw_gemm_kernel<512, 128, 128><<<dim3(NTOT / 128, 4), 256, 0, stream>>>(
      dwo_v, wt_v, sc_v, sh_v, out_val);
}
